// MLPEncoder_graph_38912403701956
// MI455X (gfx1250) — matrix operand order repaired and checked by reading, not compiled, not run on hardware
//
#include <hip/hip_runtime.h>
#include <hip/hip_bf16.h>

// ---------------------------------------------------------------------------
// MI455X (gfx1250) fused GAT encoder, wave32, WMMA bf16->f32.
// All WMMA operands fed by 2x16B contiguous loads per lane:
//   A tiles: row-major (MxK) storage, lane reads K{khalf..+7} and K{16+khalf..+7}
//   B tiles: N-major (transposed) storage, lane reads K{khalf..+7} and K{16+khalf..+7}
// P (attention probabilities) built directly in registers in A-layout,
// packed pairwise with v_perm_b32.
// ---------------------------------------------------------------------------

typedef __bf16 bf16_t;
typedef __attribute__((ext_vector_type(16))) __bf16 v16bf;
typedef __attribute__((ext_vector_type(8)))  float  v8f;

#define NTOT  2048
#define FIN   128
#define HD    128
#define OD    64
#define BATCH 8

// output tuple offsets (elements): x_proj, logits, adj_A1, eye, z, z_pos, adj_A, Wa
#define O_XPROJ  0
#define O_LOGITS 1048576
#define O_ADJA1  2097152
#define O_EYE    6291456
#define O_Z      10485760
#define O_ZPOS   10485761
#define O_ADJA   14680065
#define O_WA     18874369

// workspace offsets (bytes); total ~14.9 MB
#define WS_WB    0          // 128x128 bf16  W      [h][f]   (N-major for GEMM1 B)
#define WS_FC2B  32768      // 64x128  bf16  fc2_w  [o][h]   (N-major for GEMM3 B)
#define WS_HBT   49152      // B*H*N   bf16  h^T    [b][h][n] (N-major for GEMM2 B)
#define WS_ESRC  4243456    // B*N f32
#define WS_EDST  4308992    // B*N f32
#define WS_XPWT  4374528    // B*O*N   bf16  (x_proj+Wa)^T   (N-major for GEMM4 B)
#define WS_ADJF  6471680    // N*N     bf16  I - adj_A1^T (row-major A for GEMM4)

// ---- bf16 helpers (RNE) ---------------------------------------------------
__device__ __forceinline__ unsigned rne_hi(float f) {
  union { float f; unsigned u; } v; v.f = f;
  return v.u + 0x7FFFu + ((v.u >> 16) & 1u);   // rounded bf16 in bits [31:16]
}
// pack two floats -> two bf16 in one dword: one v_perm_b32 merges high halves
__device__ __forceinline__ unsigned pk2bf(float lo, float hi) {
  return __builtin_amdgcn_perm(rne_hi(hi), rne_hi(lo), 0x07060302u);
}
__device__ __forceinline__ bf16_t f2bf(float f) {
  const unsigned short h = (unsigned short)(rne_hi(f) >> 16);
  bf16_t out; __builtin_memcpy(&out, &h, 2);
  return out;
}

union B32x2 { uint4 q[2]; unsigned w[8]; v16bf v; };

// A operand: 16x32 bf16 tile, row-major with row stride `stride` (elements).
__device__ __forceinline__ v16bf load_A(const bf16_t* base, size_t stride) {
  const int lane = threadIdx.x & 31;
  const bf16_t* p = base + (size_t)(lane & 15) * stride + ((lane >> 4) * 8);
  B32x2 u;
  u.q[0] = *(const uint4*)(p);
  u.q[1] = *(const uint4*)(p + 16);
  return u.v;
}
// B operand: 32(K)x16(N) tile stored N-major with row stride `stride`.
__device__ __forceinline__ v16bf load_B(const bf16_t* base, size_t stride) {
  const int lane = threadIdx.x & 31;
  const bf16_t* p = base + (size_t)(lane & 15) * stride + ((lane >> 4) * 8);
  B32x2 u;
  u.q[0] = *(const uint4*)(p);
  u.q[1] = *(const uint4*)(p + 16);
  return u.v;
}
__device__ __forceinline__ v8f wmma_bf16(v16bf a, v16bf b, v8f c) {
  return __builtin_amdgcn_wmma_f32_16x16x32_bf16(false, a, false, b, (short)0, c,
                                                 false, false);
}
// pack 8 f32 -> 8 bf16 (16 bytes): 4x v_perm path
__device__ __forceinline__ uint4 pack8_bf(const v8f& a) {
  uint4 q;
  q.x = pk2bf(a[0], a[1]);
  q.y = pk2bf(a[2], a[3]);
  q.z = pk2bf(a[4], a[5]);
  q.w = pk2bf(a[6], a[7]);
  return q;
}
__device__ __forceinline__ unsigned long long pack4_bf(float4 f) {
  union { unsigned w[2]; unsigned long long q; } u;
  u.w[0] = pk2bf(f.x, f.y);
  u.w[1] = pk2bf(f.z, f.w);
  return u.q;
}

// masked leaky-relu edge score
__device__ __forceinline__ float edge_e(float esm, float ed, float ad, bool diag) {
  float ev = esm + ed;
  ev = ev > 0.f ? ev : 0.2f * ev;
  const float gate = ad + (diag ? 1.f : 0.f);
  return (gate <= 0.f) ? -1.0e9f : ev;
}

// ---------------------------------------------------------------------------
// k_prep: straight f32 -> bf16 conversion of W (h,f) and fc2_w (o,h)
// ---------------------------------------------------------------------------
__global__ __launch_bounds__(256) void k_prep(const float* __restrict__ W,
                                              const float* __restrict__ fc2w,
                                              bf16_t* __restrict__ wb,
                                              bf16_t* __restrict__ fc2b16) {
  const int i = blockIdx.x * 256 + threadIdx.x;
  if (i < FIN * HD) wb[i] = f2bf(W[i]);
  else if (i < FIN * HD + OD * HD) fc2b16[i - FIN * HD] = f2bf(fc2w[i - FIN * HD]);
}

// ---------------------------------------------------------------------------
// k_h: h = x @ W^T (WMMA), h stored transposed hbT[b][h][n] via packed b128
// stores; e_src/e_dst from accumulators via shuffle reductions. 4 waves/block.
// ---------------------------------------------------------------------------
__global__ __launch_bounds__(128) void k_h(const float* __restrict__ x,
                                           const float* __restrict__ asrc,
                                           const float* __restrict__ adst,
                                           const bf16_t* __restrict__ wb,
                                           bf16_t* __restrict__ hbT,
                                           float* __restrict__ esrc,
                                           float* __restrict__ edst) {
  __shared__ bf16_t wlds[HD * FIN];       // 32 KB  [h][f]
  __shared__ bf16_t xlds[4][16 * FIN];    // 16 KB  per-wave A tile
  const int tid = threadIdx.x, w = tid >> 5, lane = tid & 31;
  const int tile = blockIdx.x * 4 + w;
  const int b = tile >> 7;
  const int m0 = (tile & 127) << 4;

  { // cooperative stage of bf16 W (32KB) -> LDS
    const uint4* s = (const uint4*)wb;
    uint4* d = (uint4*)wlds;
    for (int i = tid; i < HD * FIN * 2 / 16; i += 128) d[i] = s[i];
  }
  { // per-wave x tile: float4 load -> 4x bf16 -> ds_store_b64
    const float4* xs = (const float4*)(x + ((size_t)b * NTOT + m0) * FIN);
    for (int i = lane; i < 16 * FIN / 4; i += 32)
      *(unsigned long long*)(&xlds[w][i * 4]) = pack4_bf(xs[i]);
  }
  __syncthreads();

  v16bf a[4];
#pragma unroll
  for (int ks = 0; ks < 4; ++ks) a[ks] = load_A(xlds[w] + ks * 32, FIN);

  const int nn = lane & 15;
  const int moff = (lane >> 4) * 8;
  float psum_s[8], psum_d[8];
#pragma unroll
  for (int r = 0; r < 8; ++r) { psum_s[r] = 0.f; psum_d[r] = 0.f; }

#pragma unroll
  for (int t = 0; t < 8; ++t) {
    v8f acc = 0.0f;
#pragma unroll
    for (int ks = 0; ks < 4; ++ks) {
      v16bf bm = load_B(wlds + (size_t)(t * 16) * FIN + ks * 32, FIN);
      acc = wmma_bf16(a[ks], bm, acc);
    }
    const int hh = t * 16 + nn;
    *(uint4*)(hbT + ((size_t)(b * HD + hh)) * NTOT + m0 + moff) = pack8_bf(acc);
    const float as = asrc[hh], ad = adst[hh];
#pragma unroll
    for (int r = 0; r < 8; ++r) { psum_s[r] += acc[r] * as; psum_d[r] += acc[r] * ad; }
  }
#pragma unroll
  for (int r = 0; r < 8; ++r) {
#pragma unroll
    for (int msk = 1; msk < 16; msk <<= 1) {
      psum_s[r] += __shfl_xor(psum_s[r], msk, 32);
      psum_d[r] += __shfl_xor(psum_d[r], msk, 32);
    }
  }
  if (nn == 0) {  // lanes 0 (rows m0..m0+7) and 16 (rows m0+8..m0+15)
#pragma unroll
    for (int r = 0; r < 8; ++r) {
      esrc[(size_t)b * NTOT + m0 + moff + r] = psum_s[r];
      edst[(size_t)b * NTOT + m0 + moff + r] = psum_d[r];
    }
  }
}

// ---------------------------------------------------------------------------
// k_adj (4-wide): adj_A1 = clip(sinh(3*clip(adj,-3,3)),±1000); eye; copies;
// adjF[m][n] = (m==n) - adj_A1[n][m] bf16. float4/b128 for all f32 traffic.
// ---------------------------------------------------------------------------
__global__ __launch_bounds__(256) void k_adj(const float* __restrict__ adjA,
                                             const float* __restrict__ zpos,
                                             float* __restrict__ out,
                                             bf16_t* __restrict__ adjF) {
  const size_t g = (size_t)blockIdx.x * 256 + threadIdx.x;  // 4 elems each
  const size_t base = g * 4;
  if (base >= (size_t)NTOT * NTOT) return;
  const int r = (int)(base >> 11), c = (int)(base & 2047);
  const float4 av = *(const float4*)(adjA + base);
  const float4 zp = *(const float4*)(zpos + base);
  float s[4];
  const float a4[4] = {av.x, av.y, av.z, av.w};
#pragma unroll
  for (int j = 0; j < 4; ++j) {
    const float t = fminf(fmaxf(a4[j], -3.f), 3.f) * 3.f;
    float sv = 0.5f * (__expf(t) - __expf(-t));
    s[j] = fminf(fmaxf(sv, -1000.f), 1000.f);
  }
  *(float4*)(out + O_ADJA1 + base) = make_float4(s[0], s[1], s[2], s[3]);
  float4 eye = make_float4(0.f, 0.f, 0.f, 0.f);
  if (r == c) eye.x = 1.f;
  else if (r == c + 1) eye.y = 1.f;
  else if (r == c + 2) eye.z = 1.f;
  else if (r == c + 3) eye.w = 1.f;
  *(float4*)(out + O_EYE + base) = eye;
  *(float4*)(out + O_ZPOS + base) = zp;
  *(float4*)(out + O_ADJA + base) = av;
#pragma unroll
  for (int j = 0; j < 4; ++j)
    adjF[(size_t)(c + j) * NTOT + r] = f2bf(((r == c + j) ? 1.f : 0.f) - s[j]);
}

__global__ __launch_bounds__(96) void k_small(const float* __restrict__ z,
                                              const float* __restrict__ Wa,
                                              float* __restrict__ out) {
  const int i = threadIdx.x;
  if (i < 64) out[O_WA + i] = Wa[i];
  else if (i == 64) out[O_Z] = z[0];
}

// ---------------------------------------------------------------------------
// k_attn: per (batch, 16-row) tile. Two-pass masked softmax, P built in
// registers in WMMA A-layout (pairwise v_perm packing), h' = P@h with B
// straight from global hbT. ELU, x_proj = h'@fc2^T + b. 8 waves/block.
// ---------------------------------------------------------------------------
__global__ __launch_bounds__(256) void k_attn(
    const float* __restrict__ adj, const float* __restrict__ esrcg,
    const float* __restrict__ edstg, const bf16_t* __restrict__ hbT,
    const bf16_t* __restrict__ fc2b16, const float* __restrict__ fc2b,
    const float* __restrict__ Wa, float* __restrict__ out,
    bf16_t* __restrict__ xpwT) {
  __shared__ float  rowsum[8][16];
  __shared__ bf16_t hact[8][16 * HD];  // 32 KB total, per-wave h_act (A tile)

  const int tid = threadIdx.x, w = tid >> 5, lane = tid & 31;
  const int tile = blockIdx.x * 8 + w;
  const int b = tile >> 7;
  const int m0 = (tile & 127) << 4;

  const int m = lane & 15;            // this lane's attention row
  const int khalf = (lane >> 4) * 8;  // A-layout K sub-offset
  const int dg = m0 + m;              // diagonal column for this row
  const float esm = esrcg[(size_t)b * NTOT + dg];
  const float* __restrict__ adjr = adj + (size_t)dg * NTOT;
  const float* __restrict__ edr = edstg + (size_t)b * NTOT;

  // ---- pass 1: masked row max ----
  float rmax;
  {
    const int n0 = (lane >> 4) << 10;
    const float4* a4 = (const float4*)(adjr + n0);
    const float4* e4 = (const float4*)(edr + n0);
    float mx = -3.0e38f;
    for (int i = 0; i < 256; ++i) {
      const float4 av = a4[i];
      const float4 ev = e4[i];
      const int nb = n0 + i * 4;
      mx = fmaxf(mx, edge_e(esm, ev.x, av.x, dg == nb));
      mx = fmaxf(mx, edge_e(esm, ev.y, av.y, dg == nb + 1));
      mx = fmaxf(mx, edge_e(esm, ev.z, av.z, dg == nb + 2));
      mx = fmaxf(mx, edge_e(esm, ev.w, av.w, dg == nb + 3));
    }
    mx = fmaxf(mx, __shfl_xor(mx, 16, 32));
    rmax = mx;
  }

  // ---- pass 2: P in registers (A-layout) + WMMA accumulate h' = P @ h ----
  v8f acc[8];
#pragma unroll
  for (int t = 0; t < 8; ++t) acc[t] = 0.0f;
  float ps_total = 0.f;

  for (int nc = 0; nc < NTOT; nc += 32) {
    const int c0 = nc + khalf;
    float edv[16], adv[16], p[16];
    *(float4*)(edv + 0)  = *(const float4*)(edr + c0);
    *(float4*)(edv + 4)  = *(const float4*)(edr + c0 + 4);
    *(float4*)(edv + 8)  = *(const float4*)(edr + c0 + 16);
    *(float4*)(edv + 12) = *(const float4*)(edr + c0 + 20);
    *(float4*)(adv + 0)  = *(const float4*)(adjr + c0);
    *(float4*)(adv + 4)  = *(const float4*)(adjr + c0 + 4);
    *(float4*)(adv + 8)  = *(const float4*)(adjr + c0 + 16);
    *(float4*)(adv + 12) = *(const float4*)(adjr + c0 + 20);
#pragma unroll
    for (int j = 0; j < 16; ++j) {
      const int col = c0 + (j < 8 ? j : j + 8);
      p[j] = __expf(edge_e(esm, edv[j], adv[j], dg == col) - rmax);
      ps_total += p[j];
    }
    B32x2 au;
#pragma unroll
    for (int j2 = 0; j2 < 8; ++j2) au.w[j2] = pk2bf(p[2 * j2], p[2 * j2 + 1]);
    const v16bf a = au.v;
#pragma unroll
    for (int t = 0; t < 8; ++t) {
      v16bf bm = load_B(hbT + (size_t)(b * HD + t * 16) * NTOT + nc, NTOT);
      acc[t] = wmma_bf16(a, bm, acc[t]);
    }
  }

  // ---- row sums: lane pair (khalf 0/8) covers the full K range ----
  {
    float rs = ps_total + __shfl_xor(ps_total, 16, 32);
    if (lane < 16) rowsum[w][lane] = rs;
  }

  // ---- normalize + ELU; transpose-stage h_act as A tile ----
  const int nn = lane & 15;
  const int moff = (lane >> 4) * 8;
  bf16_t* ha = hact[w];
#pragma unroll
  for (int t = 0; t < 8; ++t) {
#pragma unroll
    for (int r = 0; r < 8; ++r) {
      float v = acc[t][r] / rowsum[w][r + moff];
      v = v > 0.f ? v : (__expf(v) - 1.f);
      ha[(r + moff) * HD + t * 16 + nn] = f2bf(v);
    }
  }

  // ---- x_proj = h_act @ fc2^T + fc2_b ; xpwT = bf16(x_proj + Wa)^T ----
  v16bf a2[4];
#pragma unroll
  for (int ks = 0; ks < 4; ++ks) a2[ks] = load_A(ha + ks * 32, HD);
#pragma unroll
  for (int t = 0; t < 4; ++t) {
    v8f c2 = 0.0f;
#pragma unroll
    for (int ks = 0; ks < 4; ++ks) {
      v16bf bm = load_B(fc2b16 + (size_t)(t * 16) * HD + ks * 32, HD);
      c2 = wmma_bf16(a2[ks], bm, c2);
    }
    const int o = t * 16 + nn;
    const float bias = fc2b[o];
    const float wav = Wa[o];
    v8f xw;
#pragma unroll
    for (int r = 0; r < 8; ++r) {
      const float xv = c2[r] + bias;
      out[O_XPROJ + ((size_t)b * NTOT + m0 + moff + r) * OD + o] = xv;
      xw[r] = xv + wav;
    }
    *(uint4*)(xpwT + (size_t)(b * OD + o) * NTOT + m0 + moff) = pack8_bf(xw);
  }
}

// ---------------------------------------------------------------------------
// k_logits: logits[b,m,o] = sum_n adjF[m,n]*xpwT[b,o,n] - Wa[o]  (K = 2048)
// No LDS: A and B both straight from global with 2x b128 per operand.
// ---------------------------------------------------------------------------
__global__ __launch_bounds__(256) void k_logits(const bf16_t* __restrict__ adjF,
                                                const bf16_t* __restrict__ xpwT,
                                                const float* __restrict__ Wa,
                                                float* __restrict__ out) {
  const int tid = threadIdx.x, w = tid >> 5, lane = tid & 31;
  const int tile = blockIdx.x * 8 + w;
  const int b = tile >> 7;
  const int m0 = (tile & 127) << 4;

  v8f acc[4];
#pragma unroll
  for (int t = 0; t < 4; ++t) acc[t] = 0.0f;

  const bf16_t* aBase = adjF + (size_t)m0 * NTOT;
  const bf16_t* bBase = xpwT + (size_t)b * OD * NTOT;
  for (int kc = 0; kc < NTOT; kc += 32) {
    if (kc + 32 < NTOT)
      __builtin_prefetch(aBase + (size_t)(lane & 15) * NTOT + kc + 32, 0, 0);
    const v16bf a = load_A(aBase + kc, NTOT);
#pragma unroll
    for (int t = 0; t < 4; ++t) {
      v16bf bm = load_B(bBase + (size_t)(t * 16) * NTOT + kc, NTOT);
      acc[t] = wmma_bf16(a, bm, acc[t]);
    }
  }
  const int nn = lane & 15;
  const int moff = (lane >> 4) * 8;
#pragma unroll
  for (int t = 0; t < 4; ++t) {
    const int o = t * 16 + nn;
    const float wav = Wa[o];
#pragma unroll
    for (int r = 0; r < 8; ++r)
      out[O_LOGITS + ((size_t)b * NTOT + m0 + moff + r) * OD + o] = acc[t][r] - wav;
  }
}

// ---------------------------------------------------------------------------
extern "C" void kernel_launch(void* const* d_in, const int* in_sizes, int n_in,
                              void* d_out, int out_size, void* d_ws, size_t ws_size,
                              hipStream_t stream) {
  (void)in_sizes; (void)n_in; (void)out_size; (void)ws_size;
  const float* x    = (const float*)d_in[0];
  const float* adjA = (const float*)d_in[1];
  const float* W    = (const float*)d_in[2];
  const float* asrc = (const float*)d_in[3];
  const float* adst = (const float*)d_in[4];
  const float* fc2w = (const float*)d_in[5];
  const float* fc2b = (const float*)d_in[6];
  const float* Wa   = (const float*)d_in[7];
  const float* z    = (const float*)d_in[8];
  const float* zpos = (const float*)d_in[9];
  float* out = (float*)d_out;
  char* ws = (char*)d_ws;
  bf16_t* wb      = (bf16_t*)(ws + WS_WB);
  bf16_t* fc2b16  = (bf16_t*)(ws + WS_FC2B);
  bf16_t* hbT     = (bf16_t*)(ws + WS_HBT);
  float*  esrc    = (float*)(ws + WS_ESRC);
  float*  edst    = (float*)(ws + WS_EDST);
  bf16_t* xpwT    = (bf16_t*)(ws + WS_XPWT);
  bf16_t* adjF    = (bf16_t*)(ws + WS_ADJF);

  k_prep  <<<96,   256, 0, stream>>>(W, fc2w, wb, fc2b16);
  k_h     <<<256,  128, 0, stream>>>(x, asrc, adst, wb, hbT, esrc, edst);
  k_adj   <<<4096, 256, 0, stream>>>(adjA, zpos, out, adjF);
  k_attn  <<<128,  256, 0, stream>>>(adjA, esrc, edst, hbT, fc2b16, fc2b, Wa, out, xpwT);
  k_small <<<1,     96, 0, stream>>>(z, Wa, out);
  k_logits<<<128,  256, 0, stream>>>(adjF, xpwT, Wa, out);
}
